// RelativeMultiHeadAttention_64304250355918
// MI455X (gfx1250) — matrix operand order repaired and checked by reading, not compiled, not run on hardware
//
#include <hip/hip_runtime.h>

// ---------------- types ----------------
typedef _Float16 h16;
typedef __attribute__((ext_vector_type(16))) _Float16 h16x16;
typedef __attribute__((ext_vector_type(8)))  _Float16 h16x8;
typedef __attribute__((ext_vector_type(8)))  float    f32x8;
typedef int i32x4v __attribute__((vector_size(16)));   // builtin's pointee type

// ---------------- problem constants ----------------
#define D_MODEL   1024
#define NUM_HEADS 16
#define HEAD_DIM  64
#define SEQ_L     65
#define BATCH     256
#define ROWS      (BATCH * SEQ_L)   // 16640

// ---------------- gfx1250 async global->LDS path (probed via __has_builtin) ----
#if defined(__has_builtin)
#  if __has_builtin(__builtin_amdgcn_global_load_async_to_lds_b128)
#    define HAS_ASYNC_LDS 1
#  endif
#  if __has_builtin(__builtin_amdgcn_s_wait_asynccnt)
#    define HAS_WAIT_ASYNC 1
#  endif
#endif
#ifndef HAS_ASYNC_LDS
#define HAS_ASYNC_LDS 0
#endif
#ifndef HAS_WAIT_ASYNC
#define HAS_WAIT_ASYNC 0
#endif

#define AS1 __attribute__((address_space(1)))
#define AS3 __attribute__((address_space(3)))

// 16B global -> LDS copy; async (ASYNCcnt-tracked DMA) when the toolchain has it.
// Builtin signature (from hipcc diagnostic): (int4 AS1*, int4 AS3*, imm offset, imm cpol)
static __device__ inline void cp_async_b128(h16* lds_dst, const h16* gsrc) {
#if HAS_ASYNC_LDS
  __builtin_amdgcn_global_load_async_to_lds_b128(
      (AS1 i32x4v*)(unsigned long long)gsrc,
      (AS3 i32x4v*)lds_dst, 0, 0);
#else
  *(h16x8*)lds_dst = *(const h16x8*)gsrc;
#endif
}

static __device__ inline void wait_async() {
#if HAS_ASYNC_LDS
#  if HAS_WAIT_ASYNC
  __builtin_amdgcn_s_wait_asynccnt(0);
#  else
  asm volatile("s_wait_asynccnt 0" ::: "memory");
#  endif
#endif
}

static __device__ inline f32x8 fzero8() {
  f32x8 c;
#pragma unroll
  for (int i = 0; i < 8; ++i) c[i] = 0.0f;
  return c;
}

static __device__ inline f32x8 wmma_f16(h16x16 a, h16x16 b, f32x8 c) {
  // D = A(16x32 f16) * B(32x16 f16) + C(16x16 f32)
  return __builtin_amdgcn_wmma_f32_16x16x32_f16(false, a, false, b, (short)0, c,
                                                false, false);
}

// A fragment (16x32 f16) from row-major storage; base -> element (m0,k0); ld in elems.
// Layout: lanes 0-15 row M=lane hold K {0..15}; lanes 16-31 hold K {16..31}.
static __device__ inline h16x16 frag_A(const h16* base, int ld, int lane) {
  const int hf = lane >> 4, r = lane & 15;
  const h16* row = base + r * ld + (hf << 4);
  h16x8 lo = *(const h16x8*)(row);
  h16x8 hi = *(const h16x8*)(row + 8);
  h16x16 a;
#pragma unroll
  for (int i = 0; i < 8; ++i) { a[i] = lo[i]; a[i + 8] = hi[i]; }
  return a;
}

// B fragment (32x16 f16) from N-major ("B^T") storage bt[n][k]; base -> element (n0,k0).
static __device__ inline h16x16 frag_B(const h16* bt, int ld, int lane) {
  const int hf = lane >> 4, r = lane & 15;
  const h16* row = bt + r * ld + (hf << 4);
  h16x8 lo = *(const h16x8*)(row);
  h16x8 hi = *(const h16x8*)(row + 8);
  h16x16 b;
#pragma unroll
  for (int i = 0; i < 8; ++i) { b[i] = lo[i]; b[i + 8] = hi[i]; }
  return b;
}

// B fragment (32x16 f16) from ROW-major K x N storage (async-copy friendly);
// lane's column is fixed -> 16 stride-ldn scalar LDS reads.
static __device__ inline h16x16 frag_B_rm(const h16* bs, int ldn, int n0, int lane) {
  const int hf = lane >> 4, r = lane & 15;
  const h16* p = bs + (hf << 4) * ldn + n0 + r;
  h16x16 b;
#pragma unroll
  for (int i = 0; i < 16; ++i) b[i] = p[i * ldn];
  return b;
}

// ---------------- f32 -> f16 conversion ----------------
__global__ __launch_bounds__(256)
void cvt_f32_h16_kernel(const float* __restrict__ in, h16* __restrict__ out, int n) {
  int i = (blockIdx.x * 256 + threadIdx.x) * 4;
  if (i + 3 < n) {
    float4 v = *(const float4*)(in + i);
    out[i + 0] = (h16)v.x; out[i + 1] = (h16)v.y;
    out[i + 2] = (h16)v.z; out[i + 3] = (h16)v.w;
  }
}

// ---------------- tiled f16 WMMA GEMM: C[M,N] = A[M,K] * B[K,N] ----------------
// 128x128 macro-tile, double-buffered LDS fed by async global->LDS copies.
// Requires M%128==0, N%128==0, K%32==0 (true for all uses here).
template <bool OUT_F32>
__global__ __launch_bounds__(256)
void gemm_f16_kernel(const h16* __restrict__ A, const h16* __restrict__ B,
                     void* __restrict__ C, int M, int N, int K) {
  __shared__ alignas(16) h16 As[2][128][32];   // row-major A tiles (2 stages)
  __shared__ alignas(16) h16 Bs[2][32][128];   // row-major B tiles (2 stages)
  const int t = threadIdx.x;
  const int lane = t & 31, wv = t >> 5;
  const int mw = (wv >> 1) << 5;               // 4 wave-rows  * 32
  const int nw = (wv & 1) << 6;                // 2 wave-cols  * 64
  const int mB = blockIdx.x << 7, nB = blockIdx.y << 7;
  (void)M;

  f32x8 acc[2][4];
#pragma unroll
  for (int i = 0; i < 2; ++i)
#pragma unroll
    for (int j = 0; j < 4; ++j) acc[i][j] = fzero8();

  auto load_stage = [&](int st, int k0) {
    // A tile: 128x32 f16 = 512 chunks of 16B
#pragma unroll
    for (int i = 0; i < 2; ++i) {
      int id = t + (i << 8);
      int row = id >> 2, c8 = (id & 3) << 3;
      cp_async_b128(&As[st][row][c8], &A[(size_t)(mB + row) * K + k0 + c8]);
    }
    // B tile: 32x128 f16 = 512 chunks of 16B (stays row-major)
#pragma unroll
    for (int i = 0; i < 2; ++i) {
      int id = t + (i << 8);
      int kk = id >> 4, n8 = (id & 15) << 3;
      cp_async_b128(&Bs[st][kk][n8], &B[(size_t)(k0 + kk) * N + nB + n8]);
    }
  };

  load_stage(0, 0);
  wait_async();
  __syncthreads();

  for (int k0 = 0; k0 < K; k0 += 32) {
    const int cur = (k0 >> 5) & 1;
    if (k0 + 32 < K) load_stage(cur ^ 1, k0 + 32);   // overlap DMA with WMMA

    h16x16 afr[2];
#pragma unroll
    for (int mi = 0; mi < 2; ++mi)
      afr[mi] = frag_A(&As[cur][mw + (mi << 4)][0], 32, lane);
#pragma unroll
    for (int ni = 0; ni < 4; ++ni) {
      h16x16 bfr = frag_B_rm(&Bs[cur][0][0], 128, nw + (ni << 4), lane);
#pragma unroll
      for (int mi = 0; mi < 2; ++mi)
        acc[mi][ni] = wmma_f16(afr[mi], bfr, acc[mi][ni]);
    }
    wait_async();
    __syncthreads();
  }

  const int hf = lane >> 4, r = lane & 15;
#pragma unroll
  for (int mi = 0; mi < 2; ++mi)
#pragma unroll
    for (int ni = 0; ni < 4; ++ni)
#pragma unroll
      for (int i = 0; i < 8; ++i) {
        size_t m = (size_t)mB + mw + (mi << 4) + (hf << 3) + i;
        size_t n = (size_t)nB + nw + (ni << 4) + r;
        if (OUT_F32) ((float*)C)[m * N + n] = acc[mi][ni][i];
        else         ((h16*)C)[m * N + n]   = (h16)acc[mi][ni][i];
      }
}

// ---------------- fused relative multi-head attention (one block per (b,h)) ------
// Q/K/V are f16 [ROWS, D_MODEL]; head h occupies cols [h*64, h*64+64).
// Writes ctx f16 [ROWS, D_MODEL] = (attn@V + rel_out) per head.
__global__ __launch_bounds__(256)
void rel_attn_kernel(const h16* __restrict__ Q, const h16* __restrict__ Km,
                     const h16* __restrict__ Vm,
                     const float* __restrict__ relK, const float* __restrict__ relV,
                     h16* __restrict__ ctx) {
  // Segment reuse keeps total LDS at 63 KB:
  __shared__ alignas(16) unsigned char seg0[25600]; // S f32[80][80]  -> relVt h16[64][128]
  __shared__ alignas(16) unsigned char seg1[10240]; // Ks h16[80][64] -> Vt h16[64][72]
  __shared__ alignas(16) unsigned char seg2[28672]; // Qs[80][64]+relKt[144][64] -> A2 h16[80][160]

  float (*S)[80]    = (float (*)[80])seg0;
  h16 (*relVt)[128] = (h16 (*)[128])seg0;
  h16 (*Ks)[64]     = (h16 (*)[64])seg1;
  h16 (*Vt)[72]     = (h16 (*)[72])seg1;
  h16 (*Qs)[64]     = (h16 (*)[64])seg2;
  h16 (*relKt)[64]  = (h16 (*)[64])(seg2 + 80 * 64 * 2);
  h16 (*A2)[160]    = (h16 (*)[160])seg2;

  const int t = threadIdx.x, lane = t & 31, wv = t >> 5;
  const int hf = lane >> 4, lr = lane & 15;
  const int h = blockIdx.x, b = blockIdx.y;
  const size_t rowbase = ((size_t)b * SEQ_L) * D_MODEL + (size_t)h * HEAD_DIM;

  // ---- P0: zero S; async-load Qs/Ks (rows >=65 zeroed); build relKt ----
  for (int i = t; i < 80 * 80; i += 256) ((float*)seg0)[i] = 0.0f;
  for (int id = t; id < 640; id += 256) {
    int l = id >> 3, c8 = (id & 7) << 3;
    if (l < SEQ_L) {
      cp_async_b128(&Qs[l][c8], &Q[rowbase + (size_t)l * D_MODEL + c8]);
      cp_async_b128(&Ks[l][c8], &Km[rowbase + (size_t)l * D_MODEL + c8]);
    } else {
      h16x8 z;
#pragma unroll
      for (int j = 0; j < 8; ++j) z[j] = (h16)0.0f;
      *(h16x8*)&Qs[l][c8] = z;
      *(h16x8*)&Ks[l][c8] = z;
    }
  }
  for (int id = t; id < 144 * 64; id += 256) {
    int j = id >> 6, d = id & 63;
    relKt[j][d] = (j < 129) ? (h16)relK[j * 64 + d] : (h16)0.0f;
  }
  wait_async();
  __syncthreads();

  // ---- P1: E = Q @ relK^T (80x144); scatter S[l][r] = E[l][j], r = j + l - 64 ----
  for (int tile = wv; tile < 45; tile += 8) {
    int m0 = (tile / 9) << 4, n0 = (tile % 9) << 4;
    f32x8 c = fzero8();
#pragma unroll
    for (int k0 = 0; k0 < 64; k0 += 32)
      c = wmma_f16(frag_A(&Qs[m0][k0], 64, lane), frag_B(&relKt[n0][k0], 64, lane), c);
#pragma unroll
    for (int i = 0; i < 8; ++i) {
      int l = m0 + (hf << 3) + i;
      int r = n0 + lr + l - 64;                 // unique (l,r) per element
      if (r >= 0 && r < SEQ_L) S[l][r] = c[i];
    }
  }
  __syncthreads();

  // ---- P2: S += Q @ K^T ----
  for (int tile = wv; tile < 25; tile += 8) {
    int m0 = (tile / 5) << 4, n0 = (tile % 5) << 4;
    f32x8 c = fzero8();
#pragma unroll
    for (int k0 = 0; k0 < 64; k0 += 32)
      c = wmma_f16(frag_A(&Qs[m0][k0], 64, lane), frag_B(&Ks[n0][k0], 64, lane), c);
#pragma unroll
    for (int i = 0; i < 8; ++i)
      S[m0 + (hf << 3) + i][n0 + lr] += c[i];
  }
  __syncthreads();

  // ---- P3: softmax over r in [0,65) for rows l<65 (scale 1/sqrt(64)); then V^T ----
  const float scale = 0.125f;
  for (int l = wv; l < SEQ_L; l += 8) {
    float x0 = S[l][lane] * scale;
    float x1 = S[l][lane + 32] * scale;
    float x2 = (lane == 0) ? S[l][64] * scale : -3.0e38f;
    float m = fmaxf(fmaxf(x0, x1), x2);
#pragma unroll
    for (int o = 16; o > 0; o >>= 1) m = fmaxf(m, __shfl_xor(m, o, 32));
    float e0 = __expf(x0 - m), e1 = __expf(x1 - m);
    float e2 = (lane == 0) ? __expf(x2 - m) : 0.0f;
    float s = e0 + e1 + e2;
#pragma unroll
    for (int o = 16; o > 0; o >>= 1) s += __shfl_xor(s, o, 32);
    float inv = 1.0f / s;
    S[l][lane] = e0 * inv;
    S[l][lane + 32] = e1 * inv;
    if (lane == 0) S[l][64] = e2 * inv;
  }
  // Vt[d][r] = V[r][d]; cols r>=65 zeroed (Ks is dead after P2)
  for (int id = t; id < 576; id += 256) {
    int r = id >> 3, d0 = (id & 7) << 3;
    if (r < SEQ_L) {
      h16x8 v = *(const h16x8*)&Vm[rowbase + (size_t)r * D_MODEL + d0];
#pragma unroll
      for (int j = 0; j < 8; ++j) Vt[d0 + j][r] = v[j];
    } else {
#pragma unroll
      for (int j = 0; j < 8; ++j) Vt[d0 + j][r] = (h16)0.0f;
    }
  }
  __syncthreads();

  // ---- P4: build shifted attention A2[l][r-l+80] = attn[l][r] (Qs/relKt dead) ----
  for (int i = t; i < 80 * 160; i += 256) ((h16*)seg2)[i] = (h16)0.0f;
  __syncthreads();
  for (int id = t; id < SEQ_L * SEQ_L; id += 256) {
    int l = id / SEQ_L, r = id - l * SEQ_L;
    A2[l][r - l + 80] = (h16)S[l][r];
  }
  __syncthreads();
  // relVt[d][j] = relV[j][d], j<128 (S dead now)
  for (int id = t; id < 64 * 128; id += 256) {
    int d = id >> 7, j = id & 127;
    relVt[d][j] = (h16)relV[j * 64 + d];
  }
  __syncthreads();

  // ---- P5: out = attn@V + A2@relV + rank-1 (r=64 col, dist=128 row) ----
  for (int tile = wv; tile < 20; tile += 8) {
    int m0 = (tile >> 2) << 4, n0 = (tile & 3) << 4;
    f32x8 c = fzero8();
    // content part: K-chunks r0 in {0,32}; A = per-row-shifted reads of A2
#pragma unroll
    for (int r0 = 0; r0 < 64; r0 += 32) {
      h16x16 a;
      {
        int l = m0 + lr;
        const h16* row = &A2[l][0];
        int bidx = r0 + (hf << 4) - l + 80;     // in [1,143], always in-bounds
#pragma unroll
        for (int i = 0; i < 8; ++i) { a[i] = row[bidx + i]; a[i + 8] = row[bidx + 8 + i]; }
      }
      c = wmma_f16(a, frag_B(&Vt[n0][r0], 72, lane), c);
    }
    // relative part: K-chunks j0 in {0,32,64,96}, physical A2 offset +16
#pragma unroll
    for (int j0 = 0; j0 < 128; j0 += 32)
      c = wmma_f16(frag_A(&A2[m0][j0 + 16], 160, lane),
                   frag_B(&relVt[n0][j0], 128, lane), c);
    // rank-1: content column r = 64
    float v64 = (float)Vt[n0 + lr][64];
#pragma unroll
    for (int i = 0; i < 8; ++i) {
      int l = m0 + (hf << 3) + i;
      c[i] += (float)A2[l][144 - l] * v64;      // attn[l][64]
    }
    // rank-1: relative row dist=128 (only (l,r)=(0,64))
    if (m0 == 0 && hf == 0)
      c[0] += (float)A2[0][144] * relV[128 * 64 + n0 + lr];
    // store
#pragma unroll
    for (int i = 0; i < 8; ++i) {
      int l = m0 + (hf << 3) + i;
      if (l < SEQ_L) ctx[rowbase + (size_t)l * D_MODEL + n0 + lr] = (h16)c[i];
    }
  }
}

// ---------------- host launcher ----------------
// Workspace layout (bytes), total ~203 MB:
//   qb 2*NX | kb 2*NX | vb 2*NX | Qb 2*NX | Kb 2*NX | Vb 2*NX | Wq/Wk/Wv/Wo f16 (2*NW each)
//   ctx aliases qb (qb is only read by the first GEMM, which completes earlier in-stream).
extern "C" void kernel_launch(void* const* d_in, const int* in_sizes, int n_in,
                              void* d_out, int out_size, void* d_ws, size_t ws_size,
                              hipStream_t stream) {
  const float* q    = (const float*)d_in[0];
  const float* k    = (const float*)d_in[1];
  const float* v    = (const float*)d_in[2];
  const float* Wq   = (const float*)d_in[3];
  const float* Wk   = (const float*)d_in[4];
  const float* Wv   = (const float*)d_in[5];
  const float* Wo   = (const float*)d_in[6];
  const float* relK = (const float*)d_in[7];
  const float* relV = (const float*)d_in[8];

  const size_t NX = (size_t)ROWS * D_MODEL;     // 17,039,360 elems
  const size_t NW = (size_t)D_MODEL * D_MODEL;  // 1,048,576 elems

  char* w = (char*)d_ws;
  h16* qb  = (h16*)(w);
  h16* kb  = (h16*)(w + 2 * NX);
  h16* vb  = (h16*)(w + 4 * NX);
  h16* Qb  = (h16*)(w + 6 * NX);
  h16* Kb  = (h16*)(w + 8 * NX);
  h16* Vb  = (h16*)(w + 10 * NX);
  h16* Wqb = (h16*)(w + 12 * NX);
  h16* Wkb = (h16*)(w + 12 * NX + 2 * NW);
  h16* Wvb = (h16*)(w + 12 * NX + 4 * NW);
  h16* Wob = (h16*)(w + 12 * NX + 6 * NW);
  h16* ctx = qb;   // safe reuse (stream-ordered)

  dim3 blk(256);
  dim3 g1((unsigned)((NX / 4 + 255) / 256));
  cvt_f32_h16_kernel<<<g1, blk, 0, stream>>>(q, qb, (int)NX);
  cvt_f32_h16_kernel<<<g1, blk, 0, stream>>>(k, kb, (int)NX);
  cvt_f32_h16_kernel<<<g1, blk, 0, stream>>>(v, vb, (int)NX);
  dim3 g2((unsigned)((NW / 4 + 255) / 256));
  cvt_f32_h16_kernel<<<g2, blk, 0, stream>>>(Wq, Wqb, (int)NW);
  cvt_f32_h16_kernel<<<g2, blk, 0, stream>>>(Wk, Wkb, (int)NW);
  cvt_f32_h16_kernel<<<g2, blk, 0, stream>>>(Wv, Wvb, (int)NW);
  cvt_f32_h16_kernel<<<g2, blk, 0, stream>>>(Wo, Wob, (int)NW);

  dim3 gg(ROWS / 128, D_MODEL / 128);  // (130, 8)
  gemm_f16_kernel<false><<<gg, blk, 0, stream>>>(qb, Wqb, (void*)Qb, ROWS, D_MODEL, D_MODEL);
  gemm_f16_kernel<false><<<gg, blk, 0, stream>>>(kb, Wkb, (void*)Kb, ROWS, D_MODEL, D_MODEL);
  gemm_f16_kernel<false><<<gg, blk, 0, stream>>>(vb, Wvb, (void*)Vb, ROWS, D_MODEL, D_MODEL);

  rel_attn_kernel<<<dim3(NUM_HEADS, BATCH), blk, 0, stream>>>(Qb, Kb, Vb, relK, relV, ctx);

  gemm_f16_kernel<true><<<gg, blk, 0, stream>>>(ctx, Wob, d_out, ROWS, D_MODEL, D_MODEL);

  (void)in_sizes; (void)n_in; (void)out_size; (void)ws_size;
}
